// PerceiverPrunerLayer_17703855194797
// MI455X (gfx1250) — compile-verified, matrix operand order checked by reading, not run on hardware
//
#include <hip/hip_runtime.h>

// ---------------------------------------------------------------------------
// Types for CDNA5 WMMA (wave32): V_WMMA_F32_16X16X32_BF16
// ---------------------------------------------------------------------------
typedef __attribute__((ext_vector_type(16))) __bf16 v16bf;
typedef __attribute__((ext_vector_type(8)))  float  v8f;

// Problem constants (match reference)
#define B_   4
#define S_   8192
#define H_   32
#define D_   128
#define QL_  64
#define DQ_  1024
#define HD_  4096      // H_*D_
#define DK_  32        // DQ_/H_
#define T_   2048      // target_len
#define BS_  32768     // B_*S_

// ---------------------------------------------------------------------------
// CDNA5 async global->LDS copy (tracked by ASYNCcnt). 16 bytes per lane.
// The LDS operand is the raw 32-bit offset within the wave's LDS allocation,
// obtained via generic->addrspace(3) cast.
// ---------------------------------------------------------------------------
typedef __attribute__((address_space(3))) void lds_void_t;

__device__ __forceinline__ void async_copy_b128(void* lds_ptr, const void* gptr) {
  unsigned loff = (unsigned)(unsigned long long)(lds_void_t*)lds_ptr;
  asm volatile("global_load_async_to_lds_b128 %0, %1, off"
               :: "v"(loff), "v"(gptr)
               : "memory");
}

__device__ __forceinline__ void wait_async_zero() {
  asm volatile("s_wait_asynccnt 0" ::: "memory");
}

// ---------------------------------------------------------------------------
// Kernel 0: pq[64,1024] = query[64,1024] @ Wq[1024,1024]   (bf16 WMMA, f32 out)
// One wave (32 threads) per 16x16 output tile. Tiny GEMM, simplicity wins.
// ---------------------------------------------------------------------------
__global__ void __launch_bounds__(32)
pq_gemm_kernel(const float* __restrict__ query, const float* __restrict__ Wq,
               float* __restrict__ pq) {
  const int m0 = blockIdx.y * 16;          // 0..48
  const int n0 = blockIdx.x * 16;          // 0..1008
  const int lane = threadIdx.x;
  const int lrow = lane & 15;
  const int koff = (lane >> 4) << 3;       // 0 or 8
  v8f c;
  #pragma unroll
  for (int r = 0; r < 8; ++r) c[r] = 0.0f;

  for (int k0 = 0; k0 < DQ_; k0 += 32) {
    v16bf a, b;
    // A: row-major query, lane holds row m0+lrow, K runs {koff..+7, koff+16..+23}
    const float* ap = query + (size_t)(m0 + lrow) * DQ_ + k0 + koff;
    #pragma unroll
    for (int j = 0; j < 8; ++j) { a[j] = (__bf16)ap[j]; a[8 + j] = (__bf16)ap[16 + j]; }
    // B: Wq[k][n], lane holds column n0+lrow
    #pragma unroll
    for (int j = 0; j < 16; ++j) {
      int kk = (j < 8 ? j : j + 8) + koff;
      b[j] = (__bf16)Wq[(size_t)(k0 + kk) * DQ_ + n0 + lrow];
    }
    c = __builtin_amdgcn_wmma_f32_16x16x32_bf16(false, a, false, b, (short)0, c,
                                                false, false);
  }
  // C layout: VGPR r -> M = r (+8 for lanes 16..31), N = lane&15
  const int rhi = (lane >> 4) << 3;
  #pragma unroll
  for (int r = 0; r < 8; ++r)
    pq[(size_t)(m0 + rhi + r) * DQ_ + n0 + lrow] = c[r];
}

// ---------------------------------------------------------------------------
// Kernel 1: pk[32768,1024] = key2d[32768,4096] @ Wk[4096,1024]  -> bf16 out
// 274 GFLOP dominant GEMM. 256 threads (8 waves), 128x128 C tile, K-step 32.
// Double-buffered fp32 LDS staging via GLOBAL_LOAD_ASYNC_TO_LDS_B128
// (ASYNCcnt), overlapping HBM traffic with the WMMA pipe. fp32->bf16
// conversion happens at fragment build (v_cvt_pk_bf16_f32 co-executes with
// XDL WMMA). Wave grid 4(M) x 2(N); each wave: 2x4 16x16 WMMA accumulators.
// C streamed out via LDS (bf16) for coalesced 16B stores.
// ---------------------------------------------------------------------------
union PkLds {
  struct { float A[2][128 * 32]; float B[2][32 * 128]; } s;  // 64 KB
  __bf16 C[128 * 128];                                       // 32 KB
};

__device__ __forceinline__ void pk_issue_tile(const float* __restrict__ key,
                                              const float* __restrict__ Wk,
                                              float* ldsA, float* ldsB,
                                              int m0, int n0, int k0, int tid) {
  // A tile 128x32 f32: 1024 float4s, 4 per thread
  #pragma unroll
  for (int it = 0; it < 4; ++it) {
    int e  = it * 256 + tid;
    int r  = e >> 3;                    // row 0..127
    int c4 = (e & 7) * 4;               // col 0..28
    async_copy_b128(&ldsA[r * 32 + c4], key + (size_t)(m0 + r) * HD_ + k0 + c4);
  }
  // B tile 32x128 f32, kept row-major [k][n]
  #pragma unroll
  for (int it = 0; it < 4; ++it) {
    int e  = it * 256 + tid;
    int kk = e >> 5;                    // k 0..31
    int c4 = (e & 31) * 4;              // n 0..124
    async_copy_b128(&ldsB[kk * 128 + c4], Wk + (size_t)(k0 + kk) * DQ_ + n0 + c4);
  }
}

__global__ void __launch_bounds__(256)
pk_gemm_kernel(const float* __restrict__ key, const float* __restrict__ Wk,
               __bf16* __restrict__ pk) {
  __shared__ PkLds lds;
  const int m0 = blockIdx.y * 128;
  const int n0 = blockIdx.x * 128;
  const int tid  = threadIdx.x;
  const int lane = tid & 31;
  const int wave = tid >> 5;
  const int mw = (wave & 3) * 32;   // wave M offset in tile
  const int nw = (wave >> 2) * 64;  // wave N offset in tile
  const int lrow = lane & 15;
  const int koff = (lane >> 4) << 3;

  v8f acc[2][4];
  #pragma unroll
  for (int i = 0; i < 2; ++i)
    #pragma unroll
    for (int j = 0; j < 4; ++j)
      #pragma unroll
      for (int r = 0; r < 8; ++r) acc[i][j][r] = 0.0f;

  const int NK = HD_ / 32;   // 128 K-steps
  pk_issue_tile(key, Wk, lds.s.A[0], lds.s.B[0], m0, n0, 0, tid);

  for (int ks = 0; ks < NK; ++ks) {
    const int cur = ks & 1;
    wait_async_zero();      // our async copies (into buf[cur]) have landed
    __syncthreads();        // everyone's copies landed; everyone done with buf[1-cur]
    if (ks + 1 < NK)
      pk_issue_tile(key, Wk, lds.s.A[1 - cur], lds.s.B[1 - cur],
                    m0, n0, (ks + 1) * 32, tid);

    // Build bf16 fragments from the fp32 staged tiles (cvt at read).
    v16bf afrag[2], bfrag[4];
    #pragma unroll
    for (int i = 0; i < 2; ++i) {
      const float* p = &lds.s.A[cur][(mw + i * 16 + lrow) * 32 + koff];
      #pragma unroll
      for (int j = 0; j < 8; ++j) {
        afrag[i][j]     = (__bf16)p[j];
        afrag[i][8 + j] = (__bf16)p[16 + j];
      }
    }
    #pragma unroll
    for (int j4 = 0; j4 < 4; ++j4) {
      const float* p = &lds.s.B[cur][nw + j4 * 16 + lrow];
      #pragma unroll
      for (int j = 0; j < 16; ++j) {
        int kk = (j < 8 ? j : j + 8) + koff;
        bfrag[j4][j] = (__bf16)p[kk * 128];
      }
    }
    #pragma unroll
    for (int i = 0; i < 2; ++i)
      #pragma unroll
      for (int j = 0; j < 4; ++j)
        acc[i][j] = __builtin_amdgcn_wmma_f32_16x16x32_bf16(
            false, afrag[i], false, bfrag[j], (short)0, acc[i][j], false, false);
    // no trailing barrier: next iteration's wait+barrier protects both buffers
  }

  // C -> LDS (bf16, overlaps stage buffers: barrier first) -> coalesced stores
  __syncthreads();
  const int rhi = (lane >> 4) << 3;
  #pragma unroll
  for (int i = 0; i < 2; ++i)
    #pragma unroll
    for (int j = 0; j < 4; ++j)
      #pragma unroll
      for (int r = 0; r < 8; ++r)
        lds.C[(mw + i * 16 + rhi + r) * 128 + nw + j * 16 + lrow] =
            (__bf16)acc[i][j][r];
  __syncthreads();
  #pragma unroll
  for (int it = 0; it < 8; ++it) {
    int e = it * 256 + tid;        // 2048 uint4s
    int r = e >> 4;                // row (16 uint4 per 128-col row)
    int c = (e & 15) * 8;          // bf16 col
    *(uint4*)(pk + (size_t)(m0 + r) * DQ_ + n0 + c) = *(const uint4*)&lds.C[r * 128 + c];
  }
}

// ---------------------------------------------------------------------------
// Kernel 2: per (b,h): logits[64,8192] (dk=32 dots), softmax over tokens,
// mean over queries -> scores[b,h,8192]. Two passes with recompute
// (bandwidth-bound on bf16 pk; recompute is ~free). 128 blocks x 256 thr.
// ---------------------------------------------------------------------------
__global__ void __launch_bounds__(256)
scores_kernel(const float* __restrict__ pq, const __bf16* __restrict__ pk,
              float* __restrict__ scores) {
  const int b   = blockIdx.x >> 5;
  const int h   = blockIdx.x & 31;
  const int tid = threadIdx.x;
  __shared__ float  pqf[QL_][DK_ + 1];     // padded vs bank conflicts
  __shared__ __bf16 pkc[256][40];          // padded row (only [0..31] used)
  __shared__ float  red[256];
  __shared__ float  Mq[QL_], iZq[QL_];

  for (int e = tid; e < QL_ * DK_; e += 256) {
    int q = e >> 5, j = e & 31;
    pqf[q][j] = pq[(size_t)q * DQ_ + h * DK_ + j];
  }
  __syncthreads();

  const __bf16* pkh = pk + (size_t)b * S_ * DQ_ + (size_t)h * DK_;
  const int q     = tid >> 2;
  const int slice = tid & 3;
  float m = -3.0e38f, z = 0.0f;

  // Pass 1: online max/sum per query (4 slices per q, combined below)
  for (int c0 = 0; c0 < S_; c0 += 256) {
    const uint4* src = (const uint4*)(pkh + (size_t)(c0 + tid) * DQ_);
    uint4* dst = (uint4*)&pkc[tid][0];
    dst[0] = src[0]; dst[1] = src[1]; dst[2] = src[2]; dst[3] = src[3];
    __syncthreads();
    for (int s = slice; s < 256; s += 4) {
      float l = 0.0f;
      #pragma unroll
      for (int j = 0; j < DK_; ++j) l += pqf[q][j] * (float)pkc[s][j];
      float mn = fmaxf(m, l);
      z = z * __expf(m - mn) + __expf(l - mn);
      m = mn;
    }
    __syncthreads();
  }
  red[tid] = m;
  __syncthreads();
  float m4 = fmaxf(fmaxf(red[q * 4 + 0], red[q * 4 + 1]),
                   fmaxf(red[q * 4 + 2], red[q * 4 + 3]));
  __syncthreads();
  red[tid] = z * __expf(m - m4);
  __syncthreads();
  if (slice == 0) {
    float Z = red[q * 4 + 0] + red[q * 4 + 1] + red[q * 4 + 2] + red[q * 4 + 3];
    Mq[q] = m4;
    iZq[q] = 1.0f / Z;
  }
  __syncthreads();

  // Pass 2: scores[s] = (1/64) * sum_q exp(l - Mq) * iZq
  float* sout = scores + (size_t)blockIdx.x * S_;
  for (int c0 = 0; c0 < S_; c0 += 256) {
    const uint4* src = (const uint4*)(pkh + (size_t)(c0 + tid) * DQ_);
    uint4* dst = (uint4*)&pkc[tid][0];
    dst[0] = src[0]; dst[1] = src[1]; dst[2] = src[2]; dst[3] = src[3];
    __syncthreads();
    float acc = 0.0f;
    for (int qq = 0; qq < QL_; ++qq) {
      float l = 0.0f;
      #pragma unroll
      for (int j = 0; j < DK_; ++j) l += pqf[qq][j] * (float)pkc[tid][j];
      acc += __expf(l - Mq[qq]) * iZq[qq];
    }
    sout[c0 + tid] = acc * (1.0f / 64.0f);
    __syncthreads();
  }
}

// ---------------------------------------------------------------------------
// Kernel 3: per (b,h) exact top-2048 threshold via 8-bit radix select on
// order-preserving uint keys; ties taken lowest-index-first (jax top_k);
// emit selected indices in ascending order via prefix counts.
// ---------------------------------------------------------------------------
__global__ void __launch_bounds__(256)
topk_kernel(const float* __restrict__ scores, int* __restrict__ idxOut) {
  const int bh  = blockIdx.x;
  const int tid = threadIdx.x;
  __shared__ unsigned u[S_];          // 32 KB
  __shared__ unsigned hist[256];
  __shared__ unsigned bc[2];          // {prefix, remaining k}
  __shared__ unsigned tg[256], te[256];

  const float* sc = scores + (size_t)bh * S_;
  for (int e = tid; e < S_; e += 256) {
    unsigned v = __float_as_uint(sc[e]);
    u[e] = (v & 0x80000000u) ? ~v : (v | 0x80000000u);   // monotone for >
  }
  if (tid == 0) { bc[0] = 0u; bc[1] = (unsigned)T_; }
  __syncthreads();

  for (int bp = 24; bp >= 0; bp -= 8) {
    hist[tid & 255] = 0;
    __syncthreads();
    const unsigned prefix = bc[0];
    const unsigned kcur   = bc[1];
    const unsigned mask   = (bp == 24) ? 0u : (0xFFFFFFFFu << (bp + 8));
    for (int e = tid; e < S_; e += 256) {
      unsigned v = u[e];
      if ((v & mask) == prefix) atomicAdd(&hist[(v >> bp) & 255u], 1u);
    }
    __syncthreads();
    if (tid == 0) {
      unsigned cum = 0; int bin = 0;
      for (int i = 255; i >= 0; --i) {
        if (cum + hist[i] >= kcur) { bin = i; break; }
        cum += hist[i];
      }
      bc[0] = prefix | ((unsigned)bin << bp);
      bc[1] = kcur - cum;
    }
    __syncthreads();
  }
  const unsigned theta = bc[0];
  const unsigned T     = bc[1];    // #ties to keep (lowest indices)

  // per-thread counts over 32 contiguous elements, serial block scan
  unsigned g = 0, e = 0;
  const int base = tid * 32;
  #pragma unroll
  for (int i = 0; i < 32; ++i) {
    unsigned v = u[base + i];
    g += (v > theta);
    e += (v == theta);
  }
  tg[tid] = g; te[tid] = e;
  __syncthreads();
  if (tid == 0) {
    unsigned sg = 0, se = 0;
    for (int i = 0; i < 256; ++i) {
      unsigned a = tg[i], bb = te[i];
      tg[i] = sg; te[i] = se;
      sg += a; se += bb;
    }
  }
  __syncthreads();
  unsigned gB = tg[tid], eB = te[tid];
  int* out = idxOut + (size_t)bh * T_;
  for (int i = 0; i < 32; ++i) {
    unsigned v = u[base + i];
    if (v > theta) {
      out[gB + (eB < T ? eB : T)] = base + i;
      ++gB;
    } else if (v == theta) {
      if (eB < T) out[gB + eB] = base + i;
      ++eB;
    }
  }
}

// ---------------------------------------------------------------------------
// Kernel 4: gather selected K/V rows (float4 per thread). Forward scaling is
// identity (1 + p - stop_grad(p) == 1), so pure copy.
// out[0 : B*T*H*D) = selected_keys ; out[B*T*H*D : 2*B*T*H*D) = selected_values
// ---------------------------------------------------------------------------
__global__ void __launch_bounds__(256)
gather_kernel(const float* __restrict__ key, const float* __restrict__ value,
              const int* __restrict__ idx, float* __restrict__ out) {
  const size_t n  = (size_t)blockIdx.x * 256 + threadIdx.x;  // 8,388,608 total
  const int d4 = (int)(n & 31);
  const int h  = (int)((n >> 5) & 31);
  const int t  = (int)((n >> 10) & (T_ - 1));
  const int b  = (int)(n >> 21);
  const int s  = idx[((size_t)b * H_ + h) * T_ + t];
  const size_t srcOff = (((size_t)b * S_ + s) * H_ + h) * D_ + (size_t)d4 * 4;
  const size_t dstOff = (((size_t)b * T_ + t) * H_ + h) * D_ + (size_t)d4 * 4;
  const float4 kv = *(const float4*)(key + srcOff);
  const float4 vv = *(const float4*)(value + srcOff);
  *(float4*)(out + dstOff) = kv;
  *(float4*)(out + (size_t)B_ * T_ * H_ * D_ + dstOff) = vv;
}

// ---------------------------------------------------------------------------
// Workspace layout (bytes):
//   pq     f32  : 64*1024*4          =    256 KB  @ 0
//   pk     bf16 : 32768*1024*2       =     64 MB  @ 256 KB
//   scores f32  : 128*8192*4         =      4 MB  @ 256 KB + 64 MB
//   idx    i32  : 128*2048*4         =      1 MB  @ 256 KB + 68 MB
// Total ~69.25 MB.
// ---------------------------------------------------------------------------
extern "C" void kernel_launch(void* const* d_in, const int* in_sizes, int n_in,
                              void* d_out, int out_size, void* d_ws, size_t ws_size,
                              hipStream_t stream) {
  const float* key   = (const float*)d_in[0];
  const float* value = (const float*)d_in[1];
  const float* query = (const float*)d_in[2];
  const float* Wq    = (const float*)d_in[3];
  const float* Wk    = (const float*)d_in[4];
  (void)in_sizes; (void)n_in; (void)out_size; (void)ws_size;

  char*    ws     = (char*)d_ws;
  float*   pq     = (float*)ws;
  __bf16*  pk     = (__bf16*)(ws + (size_t)(256 << 10));
  float*   scores = (float*)(ws + (size_t)(256 << 10) + ((size_t)64 << 20));
  int*     idx    = (int*)(ws + (size_t)(256 << 10) + ((size_t)68 << 20));
  float*   out    = (float*)d_out;

  pq_gemm_kernel<<<dim3(DQ_ / 16, QL_ / 16), 32, 0, stream>>>(query, Wq, pq);
  pk_gemm_kernel<<<dim3(DQ_ / 128, BS_ / 128), 256, 0, stream>>>(key, Wk, pk);
  scores_kernel<<<B_ * H_, 256, 0, stream>>>(pq, pk, scores);
  topk_kernel<<<B_ * H_, 256, 0, stream>>>(scores, idx);
  gather_kernel<<<(B_ * T_ * H_ * (D_ / 4)) / 256, 256, 0, stream>>>(key, value, idx, out);
}
